// SPVCNN_if_40630390620389
// MI455X (gfx1250) — compile-verified, matrix operand order checked by reading, not run on hardware
//
#include <hip/hip_runtime.h>

// ---------------------------------------------------------------------------
// CDNA5 (gfx1250, wave32) SPVCNN forward.
// Matrix work (3D convs as implicit GEMM, MLP head) runs on
// v_wmma_f32_16x16x32_f16: one wave per 32x16 output tile (two 16x16 WMMAs
// sharing one B fragment), K chunked by 32, im2col tiles staged in LDS and
// loaded per the ISA 16-bit A/B fragment layouts.
// BN stats via ds_add_f32 (LDS) + global_atomic_add_f32.
// ---------------------------------------------------------------------------

typedef __attribute__((ext_vector_type(16))) _Float16 v16h;
typedef __attribute__((ext_vector_type(8)))  float    v8f;

__device__ __forceinline__ int kmap(int e, int g) {
    // CDNA5 16-bit 16x32 A/B fragment K index for vector element e, lane group g
    return e + ((e < 8) ? 0 : 8) + 8 * g;
}

// ---------------- implicit-GEMM 3D conv (NDHWC, DHWIO weights) --------------
__global__ __launch_bounds__(32)
void k_conv_wmma(const float* __restrict__ x, const float* __restrict__ w,
                 float* __restrict__ y, int Gin, int Gout, int Cin, int Cout,
                 int K, int stride, int pad)
{
    __shared__ _Float16 As[1024];  // 32 voxels x 32 reduction
    __shared__ _Float16 Bs[512];   // 32 reduction x 16 out-channels
    const int lane    = threadIdx.x;
    const int voxBase = blockIdx.x * 32;
    const int coBase  = blockIdx.y * 16;
    const int Vout    = Gout * Gout * Gout;
    const int Ktot    = K * K * K * Cin;
    const int nChunks = (Ktot + 31) >> 5;

    v8f acc0 = {}, acc1 = {};
    for (int ch = 0; ch < nChunks; ++ch) {
        const int kbase = ch << 5;
        if (ch + 1 < nChunks)
            __builtin_prefetch(&w[(long)(((kbase + 32) / Cin) * Cin) * Cout], 0, 1);
        // ---- A: im2col tile (32 voxels x 32 reduction) ----
        for (int e = lane; e < 1024; e += 32) {
            int m = e >> 5, k = e & 31;
            int kk = kbase + k;
            float v = 0.f;
            int vox = voxBase + m;
            if (kk < Ktot && vox < Vout) {
                int t = kk / Cin, c = kk - t * Cin;
                int kd = t / (K * K), kh = (t / K) % K, kw = t % K;
                int oz = vox % Gout; int tm = vox / Gout;
                int oy = tm % Gout;  int ox = tm / Gout;
                int ix = ox * stride - pad + kd;
                int iy = oy * stride - pad + kh;
                int iz = oz * stride - pad + kw;
                if ((unsigned)ix < (unsigned)Gin && (unsigned)iy < (unsigned)Gin &&
                    (unsigned)iz < (unsigned)Gin)
                    v = x[(((long)ix * Gin + iy) * Gin + iz) * Cin + c];
            }
            As[e] = (_Float16)v;
        }
        // ---- B: weights (32 reduction x 16 out-channels) ----
        for (int e = lane; e < 512; e += 32) {
            int k = e >> 4, n = e & 15;
            int kk = kbase + k, co = coBase + n;
            float v = 0.f;
            if (kk < Ktot && co < Cout) {
                int t = kk / Cin, c = kk - t * Cin;
                v = w[((long)t * Cin + c) * Cout + co];
            }
            Bs[e] = (_Float16)v;
        }
        __syncthreads();
        const int g = lane >> 4, ml = lane & 15;
        union { v16h v; _Float16 h[16]; } a0, a1, b;
        #pragma unroll
        for (int e = 0; e < 16; ++e) {
            int kA = kmap(e, g);
            a0.h[e] = As[(ml << 5) | kA];
            a1.h[e] = As[((ml + 16) << 5) | kA];
            b.h[e]  = Bs[(kA << 4) | ml];
        }
        acc0 = __builtin_amdgcn_wmma_f32_16x16x32_f16(
            false, a0.v, false, b.v, (short)0, acc0, false, false);
        acc1 = __builtin_amdgcn_wmma_f32_16x16x32_f16(
            false, a1.v, false, b.v, (short)0, acc1, false, false);
        __syncthreads();
    }
    const int g = lane >> 4, ml = lane & 15;
    const int co = coBase + ml;
    #pragma unroll
    for (int r = 0; r < 8; ++r) {
        int vox = voxBase + r + 8 * g;
        if (vox < Vout && co < Cout) y[(long)vox * Cout + co] = acc0[r];
        int vox1 = vox + 16;
        if (vox1 < Vout && co < Cout) y[(long)vox1 * Cout + co] = acc1[r];
    }
}

// ---------------- GEMM: Y = A(MxK) * W(KxN) + bias ---------------------------
__global__ __launch_bounds__(32)
void k_gemm_wmma(const float* __restrict__ A, const float* __restrict__ W,
                 const float* __restrict__ bias, float* __restrict__ Y,
                 int M, int Kd, int N)
{
    __shared__ _Float16 As[1024];  // 32 rows x 32 reduction
    __shared__ _Float16 Bs[512];   // 32 reduction x 16 cols
    const int lane    = threadIdx.x;
    const int rowBase = blockIdx.x * 32;
    const int colBase = blockIdx.y * 16;
    const int nChunks = (Kd + 31) >> 5;

    v8f acc0 = {}, acc1 = {};
    for (int ch = 0; ch < nChunks; ++ch) {
        const int kbase = ch << 5;
        for (int e = lane; e < 1024; e += 32) {
            int m = e >> 5, k = e & 31;
            int kk = kbase + k, row = rowBase + m;
            float v = (row < M && kk < Kd) ? A[(long)row * Kd + kk] : 0.f;
            As[e] = (_Float16)v;
        }
        for (int e = lane; e < 512; e += 32) {
            int k = e >> 4, n = e & 15;
            int kk = kbase + k, co = colBase + n;
            float v = (kk < Kd && co < N) ? W[(long)kk * N + co] : 0.f;
            Bs[e] = (_Float16)v;
        }
        __syncthreads();
        const int g = lane >> 4, ml = lane & 15;
        union { v16h v; _Float16 h[16]; } a0, a1, b;
        #pragma unroll
        for (int e = 0; e < 16; ++e) {
            int kA = kmap(e, g);
            a0.h[e] = As[(ml << 5) | kA];
            a1.h[e] = As[((ml + 16) << 5) | kA];
            b.h[e]  = Bs[(kA << 4) | ml];
        }
        acc0 = __builtin_amdgcn_wmma_f32_16x16x32_f16(
            false, a0.v, false, b.v, (short)0, acc0, false, false);
        acc1 = __builtin_amdgcn_wmma_f32_16x16x32_f16(
            false, a1.v, false, b.v, (short)0, acc1, false, false);
        __syncthreads();
    }
    const int g = lane >> 4, ml = lane & 15;
    const int co = colBase + ml;
    #pragma unroll
    for (int r = 0; r < 8; ++r) {
        int row = rowBase + r + 8 * g;
        if (row < M && co < N) Y[(long)row * N + co] = acc0[r] + bias[co];
        int row1 = row + 16;
        if (row1 < M && co < N) Y[(long)row1 * N + co] = acc1[r] + bias[co];
    }
}

// ---------------- masked per-channel BN statistics ---------------------------
// stats layout: [0..63]=sum  [64..127]=sumsq  [128]=count
__global__ void k_bn_stats(const float* __restrict__ y, const float* __restrict__ mask,
                           int V, int C, float* __restrict__ stats)
{
    __shared__ float ssum[64], ssq[64], scnt;
    int tid = threadIdx.x;
    if (tid < 64) { ssum[tid] = 0.f; ssq[tid] = 0.f; }
    if (tid == 0) scnt = 0.f;
    __syncthreads();
    long total  = (long)V * C;
    long stride = (long)gridDim.x * blockDim.x;
    for (long j = (long)blockIdx.x * blockDim.x + tid; j < total; j += stride) {
        int i = (int)(j / C); int c = (int)(j - (long)i * C);
        float m = mask ? mask[i] : 1.f;
        float v = y[j];
        atomicAdd(&ssum[c], v * m);
        atomicAdd(&ssq[c], v * v * m);
        if (c == 0) atomicAdd(&scnt, m);
    }
    __syncthreads();
    if (tid < C) {
        atomicAdd(&stats[tid], ssum[tid]);
        atomicAdd(&stats[64 + tid], ssq[tid]);
    }
    if (tid == 0) atomicAdd(&stats[128], scnt);
}

// mode 0: relu(xhat*m)   mode 1: xhat*m   mode 2: relu(xhat*m + other)*m
__global__ void k_bn_finalize(const float* __restrict__ y, const float* __restrict__ other,
                              const float* __restrict__ mask, const float* __restrict__ stats,
                              float* __restrict__ out, int V, int C, int mode)
{
    long total = (long)V * C;
    long j = (long)blockIdx.x * blockDim.x + threadIdx.x;
    if (j >= total) return;
    int i = (int)(j / C); int c = (int)(j - (long)i * C);
    float cnt = fmaxf(stats[128], 1.f);
    float mu  = stats[c] / cnt;
    float var = fmaxf(stats[64 + c] / cnt - mu * mu, 0.f);
    float xh  = (y[j] - mu) * rsqrtf(var + 1e-5f);
    float m   = mask ? mask[i] : 1.f;
    xh *= m;
    if (mode == 2) xh += other[j];
    if (mode != 1) xh = fmaxf(xh, 0.f);
    if (mode == 2) xh *= m;
    out[j] = xh;
}

// ---------------- voxelize (scatter-average) ---------------------------------
__global__ void k_vox_scatter(const int* __restrict__ ci, const float* __restrict__ feats,
                              int n, int Gs, int C, float* __restrict__ sums,
                              float* __restrict__ cnt)
{
    long j = (long)blockIdx.x * blockDim.x + threadIdx.x;
    if (j >= (long)n * C) return;
    int p = (int)(j / C); int c = (int)(j - (long)p * C);
    int idx = (ci[p * 3] * Gs + ci[p * 3 + 1]) * Gs + ci[p * 3 + 2];
    atomicAdd(&sums[(long)idx * C + c], feats[(long)p * C + c]);
    if (c == 0) atomicAdd(&cnt[idx], 1.f);
}

__global__ void k_vox_finalize(float* __restrict__ sums, const float* __restrict__ cnt,
                               const float* __restrict__ maskMul, float* __restrict__ maskOut,
                               int V, int C)
{
    int i = blockIdx.x * blockDim.x + threadIdx.x;
    if (i >= V) return;
    float ct  = cnt[i];
    float inv = 1.f / fmaxf(ct, 1.f);
    float mm  = maskMul ? maskMul[i] : 1.f;
    for (int c = 0; c < C; ++c)
        sums[(long)i * C + c] = sums[(long)i * C + c] * inv * mm;
    if (maskOut) maskOut[i] = (ct > 0.f) ? 1.f : 0.f;
}

// ---------------- 2x mask max-pool -------------------------------------------
__global__ void k_downmask(const float* __restrict__ m, float* __restrict__ mo, int Go)
{
    int i = blockIdx.x * blockDim.x + threadIdx.x;
    int Vo = Go * Go * Go;
    if (i >= Vo) return;
    int oz = i % Go; int t = i / Go; int oy = t % Go; int ox = t / Go;
    int Gi = Go * 2;
    float v = 0.f;
    #pragma unroll
    for (int d = 0; d < 8; ++d) {
        int dx = d >> 2, dy = (d >> 1) & 1, dz = d & 1;
        v = fmaxf(v, m[((long)(ox * 2 + dx) * Gi + (oy * 2 + dy)) * Gi + (oz * 2 + dz)]);
    }
    mo[i] = v;
}

// ---------------- trilinear devoxelize ---------------------------------------
__global__ void k_devox(const float* __restrict__ grid, const float* __restrict__ mask,
                        const void* __restrict__ pts, int ptsInt, int n, int Gs, int C,
                        float invStride, float* __restrict__ out, int ostride, int ooff)
{
    long j = (long)blockIdx.x * blockDim.x + threadIdx.x;
    if (j >= (long)n * C) return;
    int p = (int)(j / C); int c = (int)(j - (long)p * C);
    float px, py, pz;
    if (ptsInt) {
        const int* ip = (const int*)pts;
        px = (float)ip[p * 3]; py = (float)ip[p * 3 + 1]; pz = (float)ip[p * 3 + 2];
    } else {
        const float* fp = (const float*)pts;
        px = fp[p * 3]; py = fp[p * 3 + 1]; pz = fp[p * 3 + 2];
    }
    float ux = px * invStride, uy = py * invStride, uz = pz * invStride;
    int ix0 = (int)floorf(ux), iy0 = (int)floorf(uy), iz0 = (int)floorf(uz);
    float fx = ux - ix0, fy = uy - iy0, fz = uz - iz0;
    float acc = 0.f;
    #pragma unroll
    for (int d = 0; d < 8; ++d) {
        int dx = d >> 2, dy = (d >> 1) & 1, dz = d & 1;
        int cx = ix0 + dx, cy = iy0 + dy, cz = iz0 + dz;
        bool valid = (unsigned)cx < (unsigned)Gs && (unsigned)cy < (unsigned)Gs &&
                     (unsigned)cz < (unsigned)Gs;
        int qx = min(max(cx, 0), Gs - 1);
        int qy = min(max(cy, 0), Gs - 1);
        int qz = min(max(cz, 0), Gs - 1);
        int idx = (qx * Gs + qy) * Gs + qz;
        float wg = (dx ? fx : 1.f - fx) * (dy ? fy : 1.f - fy) * (dz ? fz : 1.f - fz);
        wg *= (valid ? 1.f : 0.f) * mask[idx];
        acc += wg * grid[(long)idx * C + c];
    }
    out[(long)p * ostride + ooff + c] = acc;
}

// ===========================================================================
extern "C" void kernel_launch(void* const* d_in, const int* in_sizes, int n_in,
                              void* d_out, int out_size, void* d_ws, size_t ws_size,
                              hipStream_t stream)
{
    const int*   pt_coords = (const int*)d_in[0];
    const float* pt_feats  = (const float*)d_in[1];
    const float* q_coords  = (const float*)d_in[2];
    int pi = 3;
    const float* ws1  = (const float*)d_in[pi++];
    const float* ws2  = (const float*)d_in[pi++];
    const float* wd1  = (const float*)d_in[pi++];
    const float* w1a1 = (const float*)d_in[pi++];
    const float* w1b1 = (const float*)d_in[pi++];
    const float* w1a2 = (const float*)d_in[pi++];
    const float* w1b2 = (const float*)d_in[pi++];
    const float* wd2  = (const float*)d_in[pi++];
    const float* w2a1 = (const float*)d_in[pi++];
    const float* w2b1 = (const float*)d_in[pi++];
    const float* w2d1 = (const float*)d_in[pi++];
    const float* w2a2 = (const float*)d_in[pi++];
    const float* w2b2 = (const float*)d_in[pi++];
    const float* wd3  = (const float*)d_in[pi++];
    const float* w3a1 = (const float*)d_in[pi++];
    const float* w3b1 = (const float*)d_in[pi++];
    const float* w3d1 = (const float*)d_in[pi++];
    const float* w3a2 = (const float*)d_in[pi++];
    const float* w3b2 = (const float*)d_in[pi++];
    const float* l0w  = (const float*)d_in[pi++];
    const float* l0b  = (const float*)d_in[pi++];
    const float* l1w  = (const float*)d_in[pi++];
    const float* l1b  = (const float*)d_in[pi++];
    const float* l2w  = (const float*)d_in[pi++];
    const float* l2b  = (const float*)d_in[pi++];

    const int N = in_sizes[0] / 3;
    const int Q = in_sizes[2] / 3;
    const int G0 = 128, G1 = 64, G2 = 32, G3 = 16;
    const long V0 = (long)G0 * G0 * G0, V1 = (long)G1 * G1 * G1;
    const long V2 = (long)G2 * G2 * G2, V3 = (long)G3 * G3 * G3;
    const int c0 = 6, c1 = 6, c2 = 12, c3 = 25;

    // ---- workspace carve-up (floats) ----
    float* wsf = (float*)d_ws;
    size_t off = 0;
    auto alloc = [&](size_t n) { size_t r = off; off += (n + 63) & ~(size_t)63; return r; };
    float* gridP = wsf + alloc(V0 * 8);        // voxel grid / x1in; later aliased as qf
    float* cntP  = wsf + alloc(V0);
    float* m0P   = wsf + alloc(V0);
    float* m1P   = wsf + alloc(V1);
    float* m2P   = wsf + alloc(V2);
    float* m3P   = wsf + alloc(V3);
    float* x0P   = wsf + alloc(V0 * c0);
    float* r0P   = wsf + alloc(V0 * c0);
    float* x1P   = wsf + alloc(V1 * c1);
    float* s10   = wsf + alloc(V1 * c1);
    float* s11   = wsf + alloc(V1 * c1);
    float* s12   = wsf + alloc(V1 * c1);
    float* s13   = wsf + alloc(V1 * c1);
    float* x2P   = wsf + alloc(V2 * c2);
    float* s20   = wsf + alloc(V2 * c2);
    float* s21   = wsf + alloc(V2 * c2);
    float* s22   = wsf + alloc(V2 * c2);
    float* s23   = wsf + alloc(V2 * c2);
    float* x3P   = wsf + alloc(V3 * c3);
    float* s30   = wsf + alloc(V3 * c3);
    float* s31   = wsf + alloc(V3 * c3);
    float* s32   = wsf + alloc(V3 * c3);
    float* s33   = wsf + alloc(V3 * c3);
    float* z0P   = wsf + alloc((size_t)N * c0);
    float* h0P   = wsf + alloc((size_t)Q * 51);
    float* h1P   = wsf + alloc((size_t)Q * 25);
    float* statsA = wsf + alloc(192);
    float* statsB = wsf + alloc(192);
    float* qfP = gridP;  // Q*49 <= V0*8, and x1in is consumed before devox
    (void)ws_size; (void)n_in; (void)out_size;

    // ---- launch helpers ----
    auto conv_bn = [&](const float* xin, const float* wgt, float* raw, float* outp,
                       const float* msk, float* st, int Gin, int Gout, int Cin, int Cout,
                       int K, int s, int pad, int mode, const float* other) {
        long Vout = (long)Gout * Gout * Gout;
        dim3 g((unsigned)((Vout + 31) / 32), (unsigned)((Cout + 15) / 16));
        k_conv_wmma<<<g, 32, 0, stream>>>(xin, wgt, raw, Gin, Gout, Cin, Cout, K, s, pad);
        (void)hipMemsetAsync(st, 0, 129 * sizeof(float), stream);
        k_bn_stats<<<1024, 256, 0, stream>>>(raw, msk, (int)Vout, Cout, st);
        long tot = Vout * Cout;
        k_bn_finalize<<<(unsigned)((tot + 255) / 256), 256, 0, stream>>>(
            raw, other, msk, st, outp, (int)Vout, Cout, mode);
    };

    auto resblock = [&](const float* xin, const float* wa, const float* wb, const float* wds,
                        float* ra, float* rb, float* rc, float* outbuf,
                        const float* msk, int Gs, int Cin, int Cout) {
        long V = (long)Gs * Gs * Gs;
        conv_bn(xin, wa, ra, rb, msk, statsA, Gs, Gs, Cin, Cout, 3, 1, 1, 0, nullptr);
        dim3 g((unsigned)((V + 31) / 32), (unsigned)((Cout + 15) / 16));
        k_conv_wmma<<<g, 32, 0, stream>>>(rb, wb, ra, Gs, Gs, Cout, Cout, 3, 1, 1);
        (void)hipMemsetAsync(statsA, 0, 129 * sizeof(float), stream);
        k_bn_stats<<<1024, 256, 0, stream>>>(ra, msk, (int)V, Cout, statsA);
        const float* idn = xin;
        if (wds) {
            conv_bn(xin, wds, rc, rc, msk, statsB, Gs, Gs, Cin, Cout, 1, 1, 0, 1, nullptr);
            idn = rc;
        }
        long tot = V * Cout;
        k_bn_finalize<<<(unsigned)((tot + 255) / 256), 256, 0, stream>>>(
            ra, idn, msk, statsA, outbuf, (int)V, Cout, 2);
    };

    auto stage = [&](const float* xin, const float* mIn, float* mOut,
                     const float* wdn, const float* wa1, const float* wb1, const float* wds1,
                     const float* wa2, const float* wb2,
                     float* b0, float* b1, float* b2, float* b3, float* xfinal,
                     int Gin, int Cin, int Cout) {
        int Go = Gin / 2; long Vo = (long)Go * Go * Go;
        k_downmask<<<(unsigned)((Vo + 255) / 256), 256, 0, stream>>>(mIn, mOut, Go);
        conv_bn(xin, wdn, b0, b1, mOut, statsA, Gin, Go, Cin, Cin, 2, 2, 0, 0, nullptr);
        resblock(b1, wa1, wb1, wds1, b0, b2, b3, b2, mOut, Go, Cin, Cout);
        resblock(b2, wa2, wb2, nullptr, b0, b1, b3, xfinal, mOut, Go, Cout, Cout);
    };

    auto lbr = [&](const float* A, const float* Wm, const float* b, float* Y,
                   int M, int Kd, int Nc, float* dst) {
        dim3 g((unsigned)((M + 31) / 32), (unsigned)((Nc + 15) / 16));
        k_gemm_wmma<<<g, 32, 0, stream>>>(A, Wm, b, Y, M, Kd, Nc);
        (void)hipMemsetAsync(statsA, 0, 129 * sizeof(float), stream);
        k_bn_stats<<<1024, 256, 0, stream>>>(Y, nullptr, M, Nc, statsA);
        long tot = (long)M * Nc;
        k_bn_finalize<<<(unsigned)((tot + 255) / 256), 256, 0, stream>>>(
            Y, nullptr, nullptr, statsA, dst, M, Nc, 0);
    };

    // ---- 1. initial voxelization (avg point feats, C=8) ----
    (void)hipMemsetAsync(gridP, 0, (size_t)V0 * 8 * sizeof(float), stream);
    (void)hipMemsetAsync(cntP, 0, (size_t)V0 * sizeof(float), stream);
    k_vox_scatter<<<(unsigned)(((long)N * 8 + 255) / 256), 256, 0, stream>>>(
        pt_coords, pt_feats, N, G0, 8, gridP, cntP);
    k_vox_finalize<<<(unsigned)((V0 + 255) / 256), 256, 0, stream>>>(
        gridP, cntP, nullptr, m0P, (int)V0, 8);

    // ---- 2. stem: two 3^3 convs + masked BN + ReLU @128^3 ----
    conv_bn(gridP, ws1, r0P, x0P, m0P, statsA, G0, G0, 8, c0, 3, 1, 1, 0, nullptr);
    conv_bn(x0P, ws2, r0P, x0P, m0P, statsA, G0, G0, c0, c0, 3, 1, 1, 0, nullptr);

    // ---- 3. voxel_to_point -> point_to_voxel (x1in = voxelize(z0) * m0) ----
    k_devox<<<(unsigned)(((long)N * c0 + 255) / 256), 256, 0, stream>>>(
        x0P, m0P, pt_coords, 1, N, G0, c0, 1.0f, z0P, c0, 0);
    (void)hipMemsetAsync(gridP, 0, (size_t)V0 * c0 * sizeof(float), stream);
    (void)hipMemsetAsync(cntP, 0, (size_t)V0 * sizeof(float), stream);
    k_vox_scatter<<<(unsigned)(((long)N * c0 + 255) / 256), 256, 0, stream>>>(
        pt_coords, z0P, N, G0, c0, gridP, cntP);
    k_vox_finalize<<<(unsigned)((V0 + 255) / 256), 256, 0, stream>>>(
        gridP, cntP, m0P, nullptr, (int)V0, c0);

    // ---- 4. three downsampling stages ----
    stage(gridP, m0P, m1P, wd1, w1a1, w1b1, nullptr, w1a2, w1b2,
          s10, s11, s12, s13, x1P, G0, c0, c1);
    stage(x1P, m1P, m2P, wd2, w2a1, w2b1, w2d1, w2a2, w2b2,
          s20, s21, s22, s23, x2P, G1, c1, c2);
    stage(x2P, m2P, m3P, wd3, w3a1, w3b1, w3d1, w3a2, w3b2,
          s30, s31, s32, s33, x3P, G2, c2, c3);

    // ---- 5. 4-scale devox of queries into concat feature [Q, 49] ----
    k_devox<<<(unsigned)(((long)Q * c0 + 255) / 256), 256, 0, stream>>>(
        x0P, m0P, q_coords, 0, Q, G0, c0, 1.0f, qfP, 49, 0);
    k_devox<<<(unsigned)(((long)Q * c1 + 255) / 256), 256, 0, stream>>>(
        x1P, m1P, q_coords, 0, Q, G1, c1, 0.5f, qfP, 49, 6);
    k_devox<<<(unsigned)(((long)Q * c2 + 255) / 256), 256, 0, stream>>>(
        x2P, m2P, q_coords, 0, Q, G2, c2, 0.25f, qfP, 49, 12);
    k_devox<<<(unsigned)(((long)Q * c3 + 255) / 256), 256, 0, stream>>>(
        x3P, m3P, q_coords, 0, Q, G3, c3, 0.125f, qfP, 49, 24);

    // ---- 6. MLP head: 49 -> 51 -> 25 -> 32, each Linear+BN+ReLU ----
    lbr(qfP, l0w, l0b, h0P, Q, 49, 51, h0P);
    lbr(h0P, l1w, l1b, h1P, Q, 51, 25, h1P);
    // final layer: raw GEMM output (Q x 32) goes to h0P (free, 51 cols), result to d_out
    lbr(h1P, l2w, l2b, h0P, Q, 25, 32, (float*)d_out);
}
